// space_att_5995774345289
// MI455X (gfx1250) — compile-verified, matrix operand order checked by reading, not run on hardware
//
#include <hip/hip_runtime.h>

// ---------------------------------------------------------------------------
// Fused space-attention for MI455X (gfx1250, wave32, WMMA).
//   x:[4,32,1024,64] -> qkv proj (32x32) -> per-(b,w) attention over H=1024,
//   dim 32 -> out = softmax(QK^T/sqrt(128)) V * sum(w_lin) + b_lin -> [B,C,H,W]
// One workgroup (256 threads = 8 waves) per (b,w) problem; 256 workgroups.
// Exploits that A- and B-fragments share the same (position,k) lane mapping:
//  * scores computed transposed (St = K Q^T) so exp(St) is already the
//    A-fragment of P -> no LDS round trip in the inner loop
//  * projections computed as W @ X^T so each lane's outputs form one
//    contiguous 32B fragment row -> two ds_store_b128 per tile
//  * biases ride in the WMMA C operand; qscale*log2(e) folded into wq/bq so
//    the online softmax runs in the exp2 domain (bare v_exp_f32, no muls)
// K/V live in LDS (128KB); only x read + out write touch HBM (~64MB total).
// ---------------------------------------------------------------------------

typedef __attribute__((ext_vector_type(16))) __bf16 v16bf;
typedef __attribute__((ext_vector_type(8)))  float  v8f;

#define B_  4
#define C_  32
#define H_  1024
#define W_  64
#define NH  8

__device__ __forceinline__ __bf16 f2bf(float f) { return (__bf16)f; }

// k-striping of 16-bit A/B fragments (ISA 7.12.2): element i of lane L maps
// to k(i) = (i>>3)*16 + (L>>4)*8 + (i&7); fragment position = L&15.
__device__ __forceinline__ int kmap(int i, int hi) { return ((i >> 3) << 4) + hi * 8 + (i & 7); }

__global__ __launch_bounds__(256, 1)
void space_att_fused(const float* __restrict__ x,
                     const float* __restrict__ wq, const float* __restrict__ bq,
                     const float* __restrict__ wk, const float* __restrict__ bk,
                     const float* __restrict__ wv, const float* __restrict__ bv,
                     const float* __restrict__ w_lin, const float* __restrict__ b_lin,
                     float* __restrict__ out)
{
    // K tiles as A-fragments for St = K Q^T  (position = j, k = c)
    __shared__ __align__(32) __bf16 sKB[64 * 32 * 16];        // 64 KB
    // V tiles as B-fragments for O = P V     (position = c, k = h)
    __shared__ __align__(32) __bf16 sVB[32 * 2 * 32 * 16];    // 64 KB

    const int prob = blockIdx.x;          // 0..255
    const int b    = prob >> 6;
    const int w    = prob & 63;
    const int tid  = threadIdx.x;
    const int wave = tid >> 5;
    const int lane = tid & 31;
    const int n    = lane & 15;           // fragment position
    const int hi   = lane >> 4;           // lane half selects k-stripe

    // linear1 over 8 identical heads collapses to a scalar affine transform
    float gl = 0.f;
    #pragma unroll
    for (int i = 0; i < NH; ++i) gl += w_lin[i];
    const float bl = b_lin[0];
    // 1/sqrt(H/NH) * log2(e): softmax runs in exp2 domain
    const float qscale = 0.08838834764831845f * 1.4426950408889634f;

    // ---- weight fragments (A/B mapping identical): Wx rows as A operand ----
    // wxF[ot]: rows o = ot*16 + n, k = c striping; wq/bq pre-scaled by qscale
    v16bf wkF[2], wvF[2], wqF[2];
    v8f   ckv[2], cqv[2], cvv[2];        // bias C-operands
    #pragma unroll
    for (int ot = 0; ot < 2; ++ot) {
        const int o = ot * 16 + n;
        union { __bf16 e[16]; v16bf v; } tk, tv, tq;
        #pragma unroll
        for (int i = 0; i < 16; ++i) {
            const int k = kmap(i, hi);
            tk.e[i] = f2bf(wk[o * C_ + k]);
            tv.e[i] = f2bf(wv[o * C_ + k]);
            tq.e[i] = f2bf(wq[o * C_ + k] * qscale);
        }
        wkF[ot] = tk.v; wvF[ot] = tv.v; wqF[ot] = tq.v;
        #pragma unroll
        for (int r = 0; r < 8; ++r) {
            // K/Q proj computed as W@X^T: D rows = o = kmap(ot*8+r,hi)
            ckv[ot][r] = bk[kmap(ot * 8 + r, hi)];
            cqv[ot][r] = bq[kmap(ot * 8 + r, hi)] * qscale;
            // V proj computed as X^T@W^T: D cols = o = ot*16+n (constant)
            cvv[ot][r] = bv[o];
        }
    }

    const float* xb = x + (size_t)b * C_ * H_ * W_ + w;

    // ========== Phase 1: K/V projection into LDS fragments (b128 stores) ====
    for (int t2 = wave; t2 < 32; t2 += 8) {
        const int h0 = t2 * 32;
        // X^T fragments for two 16-row h-tiles (position = h, k = c)
        union { __bf16 e[16]; v16bf v; } ax[2];
        #pragma unroll
        for (int t = 0; t < 2; ++t)
            #pragma unroll
            for (int i = 0; i < 16; ++i)
                ax[t].e[i] = f2bf(xb[((size_t)kmap(i, hi) * H_ + (h0 + t * 16 + n)) * W_]);

        // K = Wk @ X^T (+bk): lane = h, regs = o -> lane's 16 outputs are one
        // contiguous fragment row of sKB tile (jt = h-tile index)
        #pragma unroll
        for (int t = 0; t < 2; ++t) {
            v8f d0 = __builtin_amdgcn_wmma_f32_16x16x32_bf16(false, wkF[0], false, ax[t].v,
                                                             (short)0, ckv[0], false, false);
            v8f d1 = __builtin_amdgcn_wmma_f32_16x16x32_bf16(false, wkF[1], false, ax[t].v,
                                                             (short)0, ckv[1], false, false);
            union { __bf16 e[16]; v16bf v; } pk;
            #pragma unroll
            for (int i = 0; i < 16; ++i) pk.e[i] = f2bf((i < 8) ? d0[i] : d1[i - 8]);
            *reinterpret_cast<v16bf*>(&sKB[(((h0 >> 4) + t) * 32 + lane) * 16]) = pk.v;
        }

        // V = X^T @ Wv^T (+bv): lane = c, regs = h; two h-tiles give the full
        // 32-k stripe -> contiguous fragment row of sVB chunk (kt = t2)
        v8f dv[2][2];
        #pragma unroll
        for (int t = 0; t < 2; ++t)
            #pragma unroll
            for (int ot = 0; ot < 2; ++ot)
                dv[t][ot] = __builtin_amdgcn_wmma_f32_16x16x32_bf16(false, ax[t].v, false, wvF[ot],
                                                                    (short)0, cvv[ot], false, false);
        #pragma unroll
        for (int ot = 0; ot < 2; ++ot) {
            union { __bf16 e[16]; v16bf v; } pv;
            #pragma unroll
            for (int i = 0; i < 16; ++i) pv.e[i] = f2bf(dv[i >> 3][ot][i & 7]);
            *reinterpret_cast<v16bf*>(&sVB[((t2 * 2 + ot) * 32 + lane) * 16]) = pv.v;
        }
    }
    __syncthreads();

    // ========== Phase 2: flash attention over Q row-blocks ==================
    float* outb = out + (size_t)b * C_ * H_ * W_ + w;

    for (int qb = wave; qb < 64; qb += 8) {
        const int h0 = qb * 16;

        // prefetch next Q-block's x gather lines
        if (qb + 8 < 64) {
            const int hn = (qb + 8) * 16 + n;
            __builtin_prefetch(&xb[((size_t)(hi * 8) * H_ + hn) * W_], 0, 0);
            __builtin_prefetch(&xb[((size_t)(16 + hi * 8) * H_ + hn) * W_], 0, 0);
        }

        // Q = Wq' @ X^T (+bq'): lane = h (q row), regs = o -> lane's 16
        // outputs are directly the B-fragment of Q^T (no LDS round trip)
        union { __bf16 e[16]; v16bf v; } ax;
        #pragma unroll
        for (int i = 0; i < 16; ++i)
            ax.e[i] = f2bf(xb[((size_t)kmap(i, hi) * H_ + (h0 + n)) * W_]);
        v8f q0 = __builtin_amdgcn_wmma_f32_16x16x32_bf16(false, wqF[0], false, ax.v,
                                                         (short)0, cqv[0], false, false);
        v8f q1 = __builtin_amdgcn_wmma_f32_16x16x32_bf16(false, wqF[1], false, ax.v,
                                                         (short)0, cqv[1], false, false);
        union { __bf16 e[16]; v16bf v; } aq;
        #pragma unroll
        for (int i = 0; i < 16; ++i) aq.e[i] = f2bf((i < 8) ? q0[i] : q1[i - 8]);

        v8f   o0 = {}, o1 = {};
        float mrun = -3.0e38f, lrun = 0.f;   // per-lane: q row = lane&15

        for (int j0 = 0; j0 < H_; j0 += 32) {
            const int jt = j0 >> 4;
            const v16bf kb0 = *reinterpret_cast<const v16bf*>(&sKB[(jt * 32 + lane) * 16]);
            const v16bf kb1 = *reinterpret_cast<const v16bf*>(&sKB[((jt + 1) * 32 + lane) * 16]);
            v8f z = {};
            // St = K Q^T (log2-domain scores): lane = q row, regs = j
            v8f s0 = __builtin_amdgcn_wmma_f32_16x16x32_bf16(false, kb0, false, aq.v,
                                                             (short)0, z, false, false);
            v8f s1 = __builtin_amdgcn_wmma_f32_16x16x32_bf16(false, kb1, false, aq.v,
                                                             (short)0, z, false, false);

            // online softmax, one q-row per lane (partner lane holds other 16 j)
            float rm = fmaxf(s0[0], s1[0]);
            #pragma unroll
            for (int i = 1; i < 8; ++i) rm = fmaxf(rm, fmaxf(s0[i], s1[i]));
            rm = fmaxf(rm, __shfl_xor(rm, 16, 32));
            const float mn    = fmaxf(mrun, rm);
            const float alpha = __builtin_amdgcn_exp2f(mrun - mn);   // bare v_exp_f32
            mrun = mn;

            union { __bf16 e[16]; v16bf v; } pp;
            float rs = 0.f;
            #pragma unroll
            for (int i = 0; i < 8; ++i) {
                const float p0 = __builtin_amdgcn_exp2f(s0[i] - mn);
                const float p1 = __builtin_amdgcn_exp2f(s1[i] - mn);
                rs += p0 + p1;
                pp.e[i]     = f2bf(p0);
                pp.e[i + 8] = f2bf(p1);
            }
            rs += __shfl_xor(rs, 16, 32);
            lrun = lrun * alpha + rs;

            // broadcast alpha (row-per-lane) into O's D-layout rows, rescale
            #pragma unroll
            for (int r = 0; r < 8; ++r) {
                const float ar = __shfl(alpha, (r + hi * 8) | (lane & 16), 32);
                o0[r] *= ar;
                o1[r] *= ar;
            }

            const int   kt  = j0 >> 5;
            const v16bf vb0 = *reinterpret_cast<const v16bf*>(&sVB[((kt * 2 + 0) * 32 + lane) * 16]);
            const v16bf vb1 = *reinterpret_cast<const v16bf*>(&sVB[((kt * 2 + 1) * 32 + lane) * 16]);
            o0 = __builtin_amdgcn_wmma_f32_16x16x32_bf16(false, pp.v, false, vb0,
                                                         (short)0, o0, false, false);
            o1 = __builtin_amdgcn_wmma_f32_16x16x32_bf16(false, pp.v, false, vb1,
                                                         (short)0, o1, false, false);
        }

        // --- epilogue: normalize, affine, scatter to [B,C,H,W] ---
        #pragma unroll
        for (int r = 0; r < 8; ++r) {
            const int   h   = h0 + r + hi * 8;
            const float l   = __shfl(lrun, (r + hi * 8) | (lane & 16), 32);
            const float s   = gl / l;                 // fold linear1 weight
            outb[((size_t)(n)      * H_ + h) * W_] = o0[r] * s + bl;
            outb[((size_t)(n + 16) * H_ + h) * W_] = o1[r] * s + bl;
        }
    }
}

extern "C" void kernel_launch(void* const* d_in, const int* in_sizes, int n_in,
                              void* d_out, int out_size, void* d_ws, size_t ws_size,
                              hipStream_t stream) {
    (void)in_sizes; (void)n_in; (void)out_size; (void)d_ws; (void)ws_size;
    const float* x     = (const float*)d_in[0];
    const float* wq    = (const float*)d_in[1];
    const float* bq    = (const float*)d_in[2];
    const float* wk    = (const float*)d_in[3];
    const float* bk    = (const float*)d_in[4];
    const float* wv    = (const float*)d_in[5];
    const float* bv    = (const float*)d_in[6];
    const float* w_lin = (const float*)d_in[7];
    const float* b_lin = (const float*)d_in[8];
    float* out = (float*)d_out;

    dim3 grid(B_ * W_);   // 256 independent (b,w) attention problems
    dim3 block(256);      // 8 waves -> 64 Q row-blocks split 8-way
    space_att_fused<<<grid, block, 0, stream>>>(x, wq, bq, wk, bk, wv, bv,
                                                w_lin, b_lin, out);
}
